// GenDecoder_30296699305999
// MI455X (gfx1250) — matrix operand order repaired and checked by reading, not compiled, not run on hardware
//
#include <hip/hip_runtime.h>
#include <hip/hip_bf16.h>
#include <math.h>

// ---------------------------------------------------------------------------
// GenDecoder for MI455X (gfx1250): attention + 2-layer LSTM + output head.
// All GEMMs run on v_wmma_f32_16x16x32_bf16 with fp32 accumulate.
// Both A and B operands are kept in packed bf16 WMMA-fragment layout so the
// GEMM hot loop is loads + WMMA only (no per-tile conversion VALU chains).
// Step-invariant attention term precomputed once.
// ---------------------------------------------------------------------------

typedef __attribute__((ext_vector_type(16))) __bf16 v16bf;
typedef __attribute__((ext_vector_type(8)))  float  v8f;

static constexpr int Bb  = 32;
static constexpr int Ss  = 512;
static constexpr int Tt  = 64;
static constexpr int INd = 512;
static constexpr int Hh  = 512;
static constexpr int HCc = 4;
static constexpr int OUTd= 512;
static constexpr int K0  = INd + HCc*Hh + Hh;  // 3072 : [x_t | c_vecs | h0]
static constexpr int N0  = 4*Hh;               // 2048
static constexpr int K1  = Hh + Hh;            // 1024 : [h0_new | h1_prev]
static constexpr int KO1 = Hh + HCc*Hh;        // 2560 : [h1_new | c_vecs]

__device__ __forceinline__ __bf16 f2bf(float f) {
  unsigned u = __builtin_bit_cast(unsigned, f);
  u += 0x7fffu + ((u >> 16) & 1u);               // round-to-nearest-even
  unsigned short h = (unsigned short)(u >> 16);
  return __builtin_bit_cast(__bf16, h);
}

__device__ __forceinline__ float sigmoidf_(float x) {
  return 1.0f / (1.0f + expf(-x));
}

// A-fragment flat index for element (m, k) of an [M, Ktot] matrix.
// Layout (ISA 16-bit A 16x32): frag[((mtile*KT + kt)*32 + lane)*16 + e]
//   m = mtile*16 + (lane&15)
//   k = kt*32 + (lane>>4)*8 + (e&7) + (e>>3)*16
__device__ __forceinline__ size_t afrag_index(int m, int k, int Ktot) {
  int KT = Ktot >> 5;
  int mt = m >> 4, low = m & 15;
  int kt = k >> 5, r = k & 31;
  int lane = low + 16 * ((r >> 3) & 1);
  int e = (r & 7) + 8 * ((r >> 4) & 1);
  return (((size_t)mt * KT + kt) * 32 + lane) * 16 + e;
}

// ---------------------------------------------------------------------------
// Weight packing: fp32 [N,K] row-major -> bf16 WMMA B-fragment layout.
// B-frag (32x16 bf16): lane = N + 16*((k>>3)&1), element e = (k&7) + 8*((k>>4)&1).
// Flat index: (((ntile*KT + ktile)*32 + lane)*16 + e)
// kOff lets us concatenate multiple source matrices along K (W_ih | W_hh).
// ---------------------------------------------------------------------------
__global__ void pack_w_kernel(const float* __restrict__ src, int srcLd, int srcColOff,
                              __bf16* __restrict__ dst, int N, int kCount, int kOff, int Ktot) {
  int idx = blockIdx.x * blockDim.x + threadIdx.x;
  if (idx >= N * kCount) return;
  int n = idx / kCount, k = idx % kCount;
  float v = src[(size_t)n * srcLd + srcColOff + k];
  int kg = kOff + k;
  int nt = n >> 4, kt = kg >> 5, r = kg & 31;
  int ln = (n & 15) + 16 * ((r >> 3) & 1), e = (r & 7) + 8 * ((r >> 4) & 1);
  int KT = Ktot >> 5;
  dst[(((size_t)nt * KT + kt) * 32 + ln) * 16 + e] = f2bf(v);
}

// Activation packing: fp32 [M,K] row-major (row stride ld) -> A-frag bf16.
__global__ void pack_a_kernel(const float* __restrict__ src, int ld,
                              __bf16* __restrict__ dst, int M, int K) {
  int idx = blockIdx.x * blockDim.x + threadIdx.x;
  if (idx >= M * K) return;
  int m = idx / K, k = idx % K;
  dst[afrag_index(m, k, K)] = f2bf(src[(size_t)m * ld + k]);
}

// Row-wise concat of up to 3 strided fp32 sources -> A-frag bf16 layout.
__global__ void concat_pack_kernel(const float* s0, int ld0, int n0,
                                   const float* s1, int ld1, int n1,
                                   const float* s2, int ld2, int n2,
                                   __bf16* __restrict__ dst, int rows) {
  int tot = n0 + n1 + n2;
  int idx = blockIdx.x * blockDim.x + threadIdx.x;
  if (idx >= rows * tot) return;
  int r = idx / tot, c = idx % tot;
  float v;
  if (c < n0)           v = s0[(size_t)r * ld0 + c];
  else if (c < n0 + n1) v = s1[(size_t)r * ld1 + (c - n0)];
  else                  v = s2[(size_t)r * ld2 + (c - n0 - n1)];
  dst[afrag_index(r, c, tot)] = f2bf(v);
}

__global__ void bias_sum_kernel(const float* a0, const float* b0, float* o0,
                                const float* a1, const float* b1, float* o1, int n) {
  int i = blockIdx.x * blockDim.x + threadIdx.x;
  if (i < n) { o0[i] = a0[i] + b0[i]; o1[i] = a1[i] + b1[i]; }
}

__global__ void init_state_kernel(const float* __restrict__ fh,
                                  float* h0, float* h1, float* c0, float* c1) {
  int i = blockIdx.x * blockDim.x + threadIdx.x;
  if (i < Bb * Hh) {
    h0[i] = fh[i];
    h1[i] = fh[Bb * Hh + i];
    c0[i] = 0.0f;
    c1[i] = 0.0f;
  }
}

// ---------------------------------------------------------------------------
// WMMA GEMM: C[M,N] = act(A[M,K] @ W^T + bias)
// A and B both pre-packed bf16 fragments. One wave per (16-row tile) x
// (64-col quad); hot loop = 1 A-frag load + 4 B-frag loads + 4 WMMAs.
// M, N multiples of 16; K multiple of 32.  ACT: 0=none 1=relu 2=tanh
// PACKOUT: write result as packed bf16 A-fragment (for feeding next GEMM).
// ---------------------------------------------------------------------------
template <int ACT, bool PACKOUT>
__global__ void wmma_gemm_kernel(const __bf16* __restrict__ Ap,
                                 const __bf16* __restrict__ Bp,
                                 const float* __restrict__ bias,
                                 void* __restrict__ Cv, int ldc,
                                 int N, int K) {
  const int lane = threadIdx.x & 31;
  const int wave = threadIdx.x >> 5;
  const int n0 = (blockIdx.x * 4 + wave) * 64;
  if (n0 >= N) return;                       // wave-uniform; EXEC stays full
  const int mtile = blockIdx.y;
  const int KT = K >> 5;
  const int nt0 = n0 >> 4;

  v8f acc[4] = {v8f{}, v8f{}, v8f{}, v8f{}};

  const v16bf* abase = (const v16bf*)Ap + ((size_t)mtile * KT) * 32 + lane;
  const v16bf* bbase = (const v16bf*)Bp + lane;

  for (int kt = 0; kt < KT; ++kt) {
    v16bf af = abase[(size_t)kt * 32];
    v16bf bf0 = bbase[((size_t)(nt0 + 0) * KT + kt) * 32];
    v16bf bf1 = bbase[((size_t)(nt0 + 1) * KT + kt) * 32];
    v16bf bf2 = bbase[((size_t)(nt0 + 2) * KT + kt) * 32];
    v16bf bf3 = bbase[((size_t)(nt0 + 3) * KT + kt) * 32];

    acc[0] = __builtin_amdgcn_wmma_f32_16x16x32_bf16(false, af, false, bf0, (short)0, acc[0], false, false);
    acc[1] = __builtin_amdgcn_wmma_f32_16x16x32_bf16(false, af, false, bf1, (short)0, acc[1], false, false);
    acc[2] = __builtin_amdgcn_wmma_f32_16x16x32_bf16(false, af, false, bf2, (short)0, acc[2], false, false);
    acc[3] = __builtin_amdgcn_wmma_f32_16x16x32_bf16(false, af, false, bf3, (short)0, acc[3], false, false);
  }

  // C/D frag layout: lane&15 = N, VGPR j -> M = (lane>>4)*8 + j
  const int mrow = mtile * 16 + (lane >> 4) * 8;
  const int ncol = lane & 15;
#pragma unroll
  for (int i = 0; i < 4; ++i) {
    int n = n0 + i * 16 + ncol;
    float bv = bias ? bias[n] : 0.0f;
#pragma unroll
    for (int j = 0; j < 8; ++j) {
      float v = acc[i][j] + bv;
      if (ACT == 1) v = fmaxf(v, 0.0f);
      if (ACT == 2) v = tanhf(v);
      if (PACKOUT) {
        // result column n becomes K-index of the next GEMM's A operand
        ((__bf16*)Cv)[afrag_index(mrow + j, n, N)] = f2bf(v);
      } else {
        ((float*)Cv)[(size_t)(mrow + j) * ldc + n] = v;
      }
    }
  }
}

// ---------------------------------------------------------------------------
// Attention scorer: one wave per (b,s).
// a1[h] = relu(pre_h[b,s,h] + mask(b,s)*lastproj[b,h]);
// logits[b,s,k] = relu(sum_h a1[h]*Wa2[k,h] + ba2[k])
// ---------------------------------------------------------------------------
__global__ void attn_kernel(const float* __restrict__ pre_h,
                            const float* __restrict__ lastproj,
                            const int* __restrict__ lengths,
                            const float* __restrict__ Wa2,
                            const float* __restrict__ ba2,
                            float* __restrict__ logits) {
  int gw = (blockIdx.x * blockDim.x + threadIdx.x) >> 5;
  int lane = threadIdx.x & 31;
  int b = gw >> 9;            // / S
  int s = gw & (Ss - 1);
  float mask = (s < lengths[b]) ? 1.0f : 0.0f;
  const float* ph = pre_h + ((size_t)b * Ss + s) * Hh;
  const float* lp = lastproj + (size_t)b * Hh;
  float a0 = 0.f, a1 = 0.f, a2 = 0.f, a3 = 0.f;
#pragma unroll 4
  for (int i = 0; i < 16; ++i) {
    int h = i * 32 + lane;
    float x = ph[h] + mask * lp[h];
    x = fmaxf(x, 0.0f);
    a0 += x * Wa2[h];
    a1 += x * Wa2[Hh + h];
    a2 += x * Wa2[2 * Hh + h];
    a3 += x * Wa2[3 * Hh + h];
  }
#pragma unroll
  for (int off = 16; off > 0; off >>= 1) {
    a0 += __shfl_xor(a0, off, 32);
    a1 += __shfl_xor(a1, off, 32);
    a2 += __shfl_xor(a2, off, 32);
    a3 += __shfl_xor(a3, off, 32);
  }
  if (lane == 0) {
    float* o = logits + ((size_t)b * Ss + s) * HCc;
    o[0] = fmaxf(a0 + ba2[0], 0.0f);
    o[1] = fmaxf(a1 + ba2[1], 0.0f);
    o[2] = fmaxf(a2 + ba2[2], 0.0f);
    o[3] = fmaxf(a3 + ba2[3], 0.0f);
  }
}

// softmax over s (in place), one block of 512 per (b,k)
__global__ void softmax_kernel(float* __restrict__ logits) {
  int b = blockIdx.x >> 2, k = blockIdx.x & 3;
  int s = threadIdx.x;
  __shared__ float red[Ss];
  float v = logits[((size_t)b * Ss + s) * HCc + k];
  red[s] = v;
  __syncthreads();
  for (int off = 256; off > 0; off >>= 1) {
    if (s < off) red[s] = fmaxf(red[s], red[s + off]);
    __syncthreads();
  }
  float mx = red[0];
  __syncthreads();
  float e = expf(v - mx);
  red[s] = e;
  __syncthreads();
  for (int off = 256; off > 0; off >>= 1) {
    if (s < off) red[s] += red[s + off];
    __syncthreads();
  }
  logits[((size_t)b * Ss + s) * HCc + k] = e / red[0];
}

// c_vecs[b, k*H + h] = sum_s attw[b,s,k]*hiddens[b,s,h]; one thread per (b,h)
__global__ void cvec_kernel(const float* __restrict__ attw,
                            const float* __restrict__ hiddens,
                            float* __restrict__ c_vecs) {
  int idx = blockIdx.x * blockDim.x + threadIdx.x;
  int b = idx >> 9, h = idx & (Hh - 1);
  const float* hp = hiddens + (size_t)b * Ss * Hh + h;
  const float* ap = attw + (size_t)b * Ss * HCc;
  float a0 = 0.f, a1 = 0.f, a2 = 0.f, a3 = 0.f;
  for (int s = 0; s < Ss; ++s) {
    float hv = hp[(size_t)s * Hh];
    a0 += ap[s * 4 + 0] * hv;
    a1 += ap[s * 4 + 1] * hv;
    a2 += ap[s * 4 + 2] * hv;
    a3 += ap[s * 4 + 3] * hv;
  }
  float* o = c_vecs + (size_t)b * (HCc * Hh) + h;
  o[0] = a0; o[Hh] = a1; o[2 * Hh] = a2; o[3 * Hh] = a3;
}

// LSTM cell (PyTorch gate order i,f,g,o), in-place h/c update
__global__ void lstm_cell_kernel(const float* __restrict__ gates,
                                 float* __restrict__ h, float* __restrict__ c) {
  int idx = blockIdx.x * blockDim.x + threadIdx.x;
  if (idx >= Bb * Hh) return;
  int b = idx >> 9, j = idx & (Hh - 1);
  const float* g = gates + (size_t)b * N0;
  float ig = sigmoidf_(g[j]);
  float fg = sigmoidf_(g[Hh + j]);
  float gg = tanhf(g[2 * Hh + j]);
  float og = sigmoidf_(g[3 * Hh + j]);
  float cn = fg * c[idx] + ig * gg;
  c[idx] = cn;
  h[idx] = og * tanhf(cn);
}

// ---------------------------------------------------------------------------
// host side
// ---------------------------------------------------------------------------
static inline void gemm_launch(hipStream_t st, int act, bool packout,
                               const __bf16* Ap, const __bf16* Bp,
                               const float* bias, void* C, int ldc,
                               int M, int N, int K) {
  dim3 grid((N + 255) / 256, M / 16);
  dim3 block(128);
  if (packout) {
    if (act == 0)      wmma_gemm_kernel<0, true><<<grid, block, 0, st>>>(Ap, Bp, bias, C, ldc, N, K);
    else if (act == 1) wmma_gemm_kernel<1, true><<<grid, block, 0, st>>>(Ap, Bp, bias, C, ldc, N, K);
    else               wmma_gemm_kernel<2, true><<<grid, block, 0, st>>>(Ap, Bp, bias, C, ldc, N, K);
  } else {
    if (act == 0)      wmma_gemm_kernel<0, false><<<grid, block, 0, st>>>(Ap, Bp, bias, C, ldc, N, K);
    else if (act == 1) wmma_gemm_kernel<1, false><<<grid, block, 0, st>>>(Ap, Bp, bias, C, ldc, N, K);
    else               wmma_gemm_kernel<2, false><<<grid, block, 0, st>>>(Ap, Bp, bias, C, ldc, N, K);
  }
}

static inline void pack_launch(hipStream_t st, const float* src, int srcLd, int colOff,
                               __bf16* dst, int N, int kCount, int kOff, int Ktot) {
  int tot = N * kCount;
  pack_w_kernel<<<(tot + 255) / 256, 256, 0, st>>>(src, srcLd, colOff, dst, N, kCount, kOff, Ktot);
}

extern "C" void kernel_launch(void* const* d_in, const int* in_sizes, int n_in,
                              void* d_out, int out_size, void* d_ws, size_t ws_size,
                              hipStream_t stream) {
  const float* inputs  = (const float*)d_in[0];   // [B,T,IN]
  const int*   lengths = (const int*)d_in[1];     // [B]
  const float* fhid    = (const float*)d_in[2];   // [L,B,H]
  const float* hiddens = (const float*)d_in[3];   // [B,S,H]
  const float* Wa1  = (const float*)d_in[6];      // [H, 2H]
  const float* ba1  = (const float*)d_in[7];
  const float* Wa2  = (const float*)d_in[8];      // [HC, H]
  const float* ba2  = (const float*)d_in[9];
  const float* Wih0 = (const float*)d_in[10];     // [4H, IN+HC*H]
  const float* Whh0 = (const float*)d_in[11];     // [4H, H]
  const float* bih0 = (const float*)d_in[12];
  const float* bhh0 = (const float*)d_in[13];
  const float* Wih1 = (const float*)d_in[14];     // [4H, H]
  const float* Whh1 = (const float*)d_in[15];     // [4H, H]
  const float* bih1 = (const float*)d_in[16];
  const float* bhh1 = (const float*)d_in[17];
  const float* Wo1  = (const float*)d_in[18];     // [H, (1+HC)H]
  const float* bo1  = (const float*)d_in[19];
  const float* Wo2  = (const float*)d_in[20];     // [OUT, H]
  const float* bo2  = (const float*)d_in[21];
  float* out = (float*)d_out;                     // [B,T,OUT]

  // workspace carve-out (256-B aligned slabs)
  char* p = (char*)d_ws;
  auto carve = [&](size_t bytes) -> char* {
    char* r = p;
    p += (bytes + 255) & ~(size_t)255;
    return r;
  };
  float*  pre_h   = (float*)carve((size_t)Bb * Ss * Hh * 4);   // 33.5 MB
  float*  logits  = (float*)carve((size_t)Bb * Ss * HCc * 4);
  float*  c_vecs  = (float*)carve((size_t)Bb * HCc * Hh * 4);
  float*  h0      = (float*)carve((size_t)Bb * Hh * 4);
  float*  c0b     = (float*)carve((size_t)Bb * Hh * 4);
  float*  h1      = (float*)carve((size_t)Bb * Hh * 4);
  float*  c1b     = (float*)carve((size_t)Bb * Hh * 4);
  float*  lastp   = (float*)carve((size_t)Bb * Hh * 4);
  float*  gates   = (float*)carve((size_t)Bb * N0 * 4);
  float*  bias0   = (float*)carve((size_t)N0 * 4);
  float*  bias1   = (float*)carve((size_t)N0 * 4);
  // packed bf16 weight (B) fragments
  __bf16* pWa1h   = (__bf16*)carve((size_t)Hh * Hh * 2);
  __bf16* pWa1d   = (__bf16*)carve((size_t)Hh * Hh * 2);
  __bf16* pW0     = (__bf16*)carve((size_t)N0 * K0 * 2);       // 12.6 MB
  __bf16* pW1     = (__bf16*)carve((size_t)N0 * K1 * 2);
  __bf16* pWo1    = (__bf16*)carve((size_t)Hh * KO1 * 2);
  __bf16* pWo2    = (__bf16*)carve((size_t)OUTd * Hh * 2);
  // packed bf16 activation (A) fragments
  __bf16* aHid    = (__bf16*)carve((size_t)Bb * Ss * Hh * 2);  // 16.8 MB
  __bf16* aH1     = (__bf16*)carve((size_t)Bb * Hh * 2);
  __bf16* aX0     = (__bf16*)carve((size_t)Bb * K0 * 2);
  __bf16* aX1     = (__bf16*)carve((size_t)Bb * K1 * 2);
  __bf16* aOcat   = (__bf16*)carve((size_t)Bb * KO1 * 2);
  __bf16* aR1     = (__bf16*)carve((size_t)Bb * Hh * 2);

  // ---- prep (runs fresh every call; deterministic) ----
  pack_launch(stream, Wa1, 2 * Hh, Hh, pWa1h, Hh, Hh, 0, Hh);      // Wa1[:, H:]
  pack_launch(stream, Wa1, 2 * Hh, 0,  pWa1d, Hh, Hh, 0, Hh);      // Wa1[:, :H]
  pack_launch(stream, Wih0, K0 - Hh, 0, pW0, N0, K0 - Hh, 0, K0);  // [W_ih0 |
  pack_launch(stream, Whh0, Hh, 0,      pW0, N0, Hh, K0 - Hh, K0); //  W_hh0]
  pack_launch(stream, Wih1, Hh, 0, pW1, N0, Hh, 0, K1);
  pack_launch(stream, Whh1, Hh, 0, pW1, N0, Hh, Hh, K1);
  pack_launch(stream, Wo1, KO1, 0, pWo1, Hh, KO1, 0, KO1);
  pack_launch(stream, Wo2, Hh, 0, pWo2, OUTd, Hh, 0, Hh);
  bias_sum_kernel<<<(N0 + 255) / 256, 256, 0, stream>>>(bih0, bhh0, bias0, bih1, bhh1, bias1, N0);
  init_state_kernel<<<(Bb * Hh + 255) / 256, 256, 0, stream>>>(fhid, h0, h1, c0b, c1b);

  // hiddens -> A fragments (once), then step-invariant attention term:
  // pre_h = hiddens @ Wa1[:,H:].T + ba1
  pack_a_kernel<<<(Bb * Ss * Hh + 255) / 256, 256, 0, stream>>>(hiddens, Hh, aHid, Bb * Ss, Hh);
  gemm_launch(stream, 0, false, aHid, pWa1h, ba1, pre_h, Hh, Bb * Ss, Hh, Hh);

  // ---- 64 sequential decode steps ----
  for (int t = 0; t < Tt; ++t) {
    // last-hidden projection (rank-1-in-s part of attention layer 1)
    pack_a_kernel<<<(Bb * Hh + 255) / 256, 256, 0, stream>>>(h1, Hh, aH1, Bb, Hh);
    gemm_launch(stream, 0, false, aH1, pWa1d, nullptr, lastp, Hh, Bb, Hh, Hh);
    attn_kernel<<<(Bb * Ss * 32) / 256, 256, 0, stream>>>(pre_h, lastp, lengths, Wa2, ba2, logits);
    softmax_kernel<<<Bb * HCc, Ss, 0, stream>>>(logits);
    cvec_kernel<<<(Bb * Hh) / 256, 256, 0, stream>>>(logits, hiddens, c_vecs);

    // LSTM layer 0: gates = [x_t | c_vecs | h0] @ [W_ih0|W_hh0]^T + b
    concat_pack_kernel<<<(Bb * K0 + 255) / 256, 256, 0, stream>>>(
        inputs + (size_t)t * INd, Tt * INd, INd,
        c_vecs, HCc * Hh, HCc * Hh,
        h0, Hh, Hh, aX0, Bb);
    gemm_launch(stream, 0, false, aX0, pW0, bias0, gates, N0, Bb, N0, K0);
    lstm_cell_kernel<<<(Bb * Hh + 255) / 256, 256, 0, stream>>>(gates, h0, c0b);

    // LSTM layer 1: gates = [h0_new | h1_prev] @ [W_ih1|W_hh1]^T + b
    concat_pack_kernel<<<(Bb * K1 + 255) / 256, 256, 0, stream>>>(
        h0, Hh, Hh, h1, Hh, Hh, h1, Hh, 0, aX1, Bb);
    gemm_launch(stream, 0, false, aX1, pW1, bias1, gates, N0, Bb, N0, K1);
    lstm_cell_kernel<<<(Bb * Hh + 255) / 256, 256, 0, stream>>>(gates, h1, c1b);

    // output head: tanh(relu([h1|c_vecs]@Wo1^T+bo1)@Wo2^T+bo2) -> out[:,t,:]
    concat_pack_kernel<<<(Bb * KO1 + 255) / 256, 256, 0, stream>>>(
        h1, Hh, Hh, c_vecs, HCc * Hh, HCc * Hh, h1, Hh, 0, aOcat, Bb);
    gemm_launch(stream, 1, true, aOcat, pWo1, bo1, aR1, 0, Bb, Hh, KO1);  // relu, packed out
    gemm_launch(stream, 2, false, aR1, pWo2, bo2, out + (size_t)t * OUTd, Tt * OUTd, Bb, OUTd, Hh);
  }
}
